// RetinaHead_9809705304074
// MI455X (gfx1250) — matrix operand order repaired and checked by reading, not compiled, not run on hardware
//
#include <hip/hip_runtime.h>

typedef __attribute__((ext_vector_type(16))) __bf16        v16bf;
typedef __attribute__((ext_vector_type(8)))  float         v8f;
typedef __attribute__((ext_vector_type(8)))  unsigned int  v8u;
typedef __attribute__((ext_vector_type(4)))  unsigned int  v4u;

static __device__ __forceinline__ unsigned short f2bf(float f) {
    unsigned int u = __float_as_uint(f);
    u += 0x7FFFu + ((u >> 16) & 1u);          // round-to-nearest-even
    return (unsigned short)(u >> 16);
}
static __device__ __forceinline__ float bf2f(unsigned short v) {
    return __uint_as_float(((unsigned int)v) << 16);
}

// CDNA5 async global->LDS copy (ASYNCcnt). ldsAddr = LDS byte offset.
static __device__ __forceinline__ void asyncLoadB32(unsigned ldsAddr, const void* g) {
    asm volatile("global_load_async_to_lds_b32 %0, %1, off"
                 :: "v"(ldsAddr), "v"(g) : "memory");
}
static __device__ __forceinline__ void waitAsync() {
    asm volatile("s_wait_asynccnt 0" ::: "memory");
}

// ---------------------------------------------------------------------------
// Weight repack: fp32 OIHW [Cout][256][3][3] -> bf16 fragment-order layout.
// For N-tile nt, K-chunk c (K=32), lane (h,m): 16 contiguous bf16 in this code's B
// fragment order (K = c*32 + 16h + 2p + w, N = nt*16 + m):
//   off = (nt*72 + c)*512 + (h*16 + m)*16 + p*2 + w        (zero-pad n>=Cout)
// ---------------------------------------------------------------------------
__global__ void wconv_kernel(const float* __restrict__ w, unsigned short* __restrict__ w2,
                             int Cout, int CoutAlloc)
{
    int tid = blockIdx.x * blockDim.x + threadIdx.x;
    int total = CoutAlloc * 2304;
    if (tid >= total) return;
    int k  = tid / CoutAlloc;        // 0..2303
    int ng = tid % CoutAlloc;        // padded output channel
    int c  = k >> 5;                 // K-chunk
    int kr = k & 31;
    int hh = kr >> 4;
    int p  = (kr & 15) >> 1;
    int wp = kr & 1;
    int nt = ng >> 4, m = ng & 15;
    float val = 0.0f;
    if (ng < Cout) {
        int tap = k >> 8;            // ky*3+kx
        int ci  = k & 255;
        val = w[((size_t)ng * 256 + ci) * 9 + tap];
    }
    w2[(size_t)(nt * 72 + c) * 512 + (hh * 16 + m) * 16 + p * 2 + wp] = f2bf(val);
}

// fp32 NCHW -> bf16 NHWC with zero halo: out = [8][H+2][W+2][256]
__global__ void cvt_kernel(const float* __restrict__ in, unsigned short* __restrict__ out,
                           int H, int W)
{
    int tid = blockIdx.x * blockDim.x + threadIdx.x;
    int Hp = H + 2, Wp = W + 2;
    int total = 8 * Hp * Wp * 256;
    if (tid >= total) return;
    int c = tid & 255;
    int cell = tid >> 8;
    int px = cell % Wp;
    int t = cell / Wp;
    int py = t % Hp;
    int b = t / Hp;
    float val = 0.0f;
    if (px >= 1 && px <= W && py >= 1 && py <= H) {
        val = in[(((size_t)b * 256 + c) * H + (py - 1)) * W + (px - 1)];
    }
    out[tid] = f2bf(val);
}

// uint4 zero fill (count must be a multiple of 8 ushorts)
__global__ void zero_kernel(unsigned int* __restrict__ p, int ndw4)
{
    int tid = blockIdx.x * blockDim.x + threadIdx.x;
    if (tid < ndw4) ((v4u*)p)[tid] = (v4u){0u, 0u, 0u, 0u};
}

// ---------------------------------------------------------------------------
// conv3x3 implicit GEMM, v_wmma_f32_16x16x32_bf16.
// Block = 256 thr (8 waves). Block tile: M=64 positions x N=256 channels.
// Wave w: N-tiles (blockIdx.y*16 + 2w, +2w+1) x M-subtiles {0..3} -> 8 WMMA/iter.
// Input is zero-padded NHWC [8][H+2][W+2][256]; im2col dwords staged into
// fragment-order LDS slots with global_load_async_to_lds_b32 (double-buffered).
// ---------------------------------------------------------------------------
__global__ __launch_bounds__(256)
void conv3x3_wmma(const unsigned short* __restrict__ in,   // padded NHWC bf16
                  const unsigned short* __restrict__ w2,   // fragment-order weights
                  const float* __restrict__ bias,          // [Cout]
                  unsigned short* __restrict__ out,        // padded 256ch or flat [pos][Cout]
                  int H, int W, int logW, int Cout, int relu, int padOut)
{
    __shared__ unsigned int AtileDW[2 * 1536];  // 2 bufs x 4 subtiles x 32 slots x 12 dw

    const int Wp = W + 2, Hp = H + 2;
    const int tid  = threadIdx.x;
    const int wave = tid >> 5;
    const int lane = tid & 31;
    const int h    = lane >> 4;
    const int m    = lane & 15;

    const int m0  = blockIdx.x * 64;
    const int nt0 = blockIdx.y * 16 + wave * 2;
    const int nt1 = nt0 + 1;
    const int n0  = nt0 * 16 + m;
    const int n1  = nt1 * 16 + m;

    // ---- im2col loader mapping: thread t -> rows lr+{0,16,32,48}, dword pair lj
    const int lr  = tid >> 4;                       // 0..15
    const int lj  = tid & 15;                       // pair index in chunk
    const int lh  = lj >> 3;                        // fragment half
    const int lpp = lj & 7;                         // fragment dword p
    const int st  = (lh * 16 + lr) * 12 + lpp;      // subtile-0 slot (dwords)
    const int cj  = lj << 1;                        // channel offset of dword

    const unsigned ldsBase = (unsigned)(size_t)AtileDW;   // LDS byte offset
    const unsigned ldsA0 = ldsBase + (unsigned)(st * 4);  // subtile s: +s*1536B
    const unsigned ldsA1 = ldsA0 + 1536u;
    const unsigned ldsA2 = ldsA0 + 3072u;
    const unsigned ldsA3 = ldsA0 + 4608u;

    // per-row padded-interior base pointers (center tap), channel dword folded in
    const unsigned short* base[4];
#pragma unroll
    for (int s = 0; s < 4; ++s) {
        int pos = m0 + lr + s * 16;
        int lx = pos & (W - 1), ly = (pos >> logW) & (H - 1), lb = pos >> (2 * logW);
        base[s] = in + ((((size_t)lb * Hp + ly + 1) * Wp) + lx + 1) * 256 + cj;
    }

    float b0v = (n0 < Cout) ? bias[n0] : 0.0f;
    float b1v = (n1 < Cout) ? bias[n1] : 0.0f;
    v8f acc0[4], acc1[4];
#pragma unroll
    for (int s = 0; s < 4; ++s)
#pragma unroll
        for (int r = 0; r < 8; ++r) { acc0[s][r] = b0v; acc1[s][r] = b1v; }

    const unsigned short* pB0 = w2 + (size_t)nt0 * 36864 + lane * 16;
    const unsigned short* pB1 = w2 + (size_t)nt1 * 36864 + lane * 16;

    // running async-source pointers; reset at each tap, advance 32 elem/chunk
    const unsigned short *q0, *q1, *q2, *q3;
    auto setTap = [&](int tap) {
        int dyq = (tap * 0x56) >> 8;                // tap / 3
        int toff = ((dyq - 1) * Wp + (tap - dyq * 3 - 1)) * 256;
        q0 = base[0] + toff; q1 = base[1] + toff;
        q2 = base[2] + toff; q3 = base[3] + toff;
    };
    auto issueChunk = [&](unsigned bo) {
        asyncLoadB32(ldsA0 + bo, q0);
        asyncLoadB32(ldsA1 + bo, q1);
        asyncLoadB32(ldsA2 + bo, q2);
        asyncLoadB32(ldsA3 + bo, q3);
        q0 += 32; q1 += 32; q2 += 32; q3 += 32;
    };

    // prologue: chunk 0 (tap 0) -> buffer 0, via async DMA
    setTap(0);
    issueChunk(0u);
    waitAsync();
    __syncthreads();

#pragma unroll 2
    for (int c = 0; c < 72; ++c) {
        // stage next chunk into the other buffer (async, overlaps WMMA)
        const int nb = c + 1;
        if (nb < 72) {
            if ((nb & 7) == 0) setTap(nb >> 3);
            issueChunk((nb & 1) ? 6144u : 0u);
        }

        // B fragments: 2 x global_load_b128 each (contiguous per lane)
        const v4u* bp0 = (const v4u*)pB0;
        const v4u* bp1 = (const v4u*)pB1;
        v4u b0lo = bp0[0], b0hi = bp0[1];
        v4u b1lo = bp1[0], b1hi = bp1[1];
        pB0 += 512; pB1 += 512;
        v16bf b0 = __builtin_bit_cast(v16bf, __builtin_shufflevector(b0lo, b0hi, 0,1,2,3,4,5,6,7));
        v16bf b1 = __builtin_bit_cast(v16bf, __builtin_shufflevector(b1lo, b1hi, 0,1,2,3,4,5,6,7));

        const int ab = ((c & 1) ? 1536 : 0) + lane * 12;
#pragma unroll
        for (int s = 0; s < 4; ++s) {
            // A fragment: 2 x 16B LDS loads (fragment-order slot)
            v4u alo = *(const v4u*)&AtileDW[ab + s * 384];
            v4u ahi = *(const v4u*)&AtileDW[ab + s * 384 + 4];
            v16bf a = __builtin_bit_cast(v16bf,
                        __builtin_shufflevector(alo, ahi, 0,1,2,3,4,5,6,7));
            acc0[s] = __builtin_amdgcn_wmma_f32_16x16x32_bf16(false, a, false, b0,
                                                              (short)0, acc0[s], false, false);
            acc1[s] = __builtin_amdgcn_wmma_f32_16x16x32_bf16(false, a, false, b1,
                                                              (short)0, acc1[s], false, false);
        }

        waitAsync();            // our async writes for next chunk are in LDS
        __syncthreads();
    }

    // C/D layout: VGPR r -> M = r + 8*h, N = m
#pragma unroll
    for (int s = 0; s < 4; ++s) {
#pragma unroll
        for (int r = 0; r < 8; ++r) {
            const int pos = m0 + s * 16 + r + h * 8;
            float x0 = acc0[s][r], x1 = acc1[s][r];
            if (relu) { x0 = fmaxf(x0, 0.0f); x1 = fmaxf(x1, 0.0f); }
            if (padOut) {
                int x = pos & (W - 1), y = (pos >> logW) & (H - 1), b = pos >> (2 * logW);
                size_t i = ((((size_t)b * Hp + y + 1) * Wp) + x + 1) * 256;
                out[i + n0] = f2bf(x0);
                out[i + n1] = f2bf(x1);
            } else {
                if (n0 < Cout) out[(size_t)pos * Cout + n0] = f2bf(x0);
                if (n1 < Cout) out[(size_t)pos * Cout + n1] = f2bf(x1);
            }
        }
    }
}

// ---------------------------------------------------------------------------
// Per (b, pos, anchor): max/argmax over 80 classes + sigmoid + anchor decode.
// Output layout: scores [8][Atot] | class_id [8][Atot] | boxes [8][Atot][4]
// ---------------------------------------------------------------------------
__global__ void postproc_kernel(const unsigned short* __restrict__ clsO, // [B*HW][720]
                                const unsigned short* __restrict__ regO, // [B*HW][36]
                                float* __restrict__ out,
                                int H, int W, int levelBase, float stride, int Atot)
{
    const int B = 8;
    int tid = blockIdx.x * blockDim.x + threadIdx.x;
    int HW = H * W;
    int total = B * HW * 9;
    if (tid >= total) return;
    int ai  = tid % 9;
    int pos = (tid / 9) % HW;
    int b   = tid / (9 * HW);

    const unsigned short* cp = clsO + ((size_t)b * HW + pos) * 720 + ai * 80;
    float best = -1e30f; int bid = 0;
    for (int c = 0; c < 80; ++c) {
        float v = bf2f(cp[c]);
        if (v > best) { best = v; bid = c; }
    }
    float score = 1.0f / (1.0f + __expf(-best));   // sigmoid is monotone

    const unsigned short* rp = regO + ((size_t)b * HW + pos) * 36 + ai * 4;
    float d0 = bf2f(rp[0]), d1 = bf2f(rp[1]), d2 = bf2f(rp[2]), d3 = bf2f(rp[3]);

    int x = pos % W, y = pos / W;
    float cx = (x + 0.5f) * stride;
    float cy = (y + 0.5f) * stride;
    const float SC[3] = {1.0f, 1.2599210499f, 1.5874010520f}; // 2^(k/3)
    const float RT[3] = {0.5f, 1.0f, 2.0f};
    float sc = 4.0f * stride * SC[ai % 3];
    float rt = RT[ai / 3];
    float aw = sc * rsqrtf(rt);   // sqrt(area/ratio)
    float ah = aw * rt;

    float pcx = cx + d0 * 0.1f * aw;
    float pcy = cy + d1 * 0.1f * ah;
    float pw  = __expf(d2 * 0.2f) * aw;
    float ph  = __expf(d3 * 0.2f) * ah;

    int aidx = levelBase + pos * 9 + ai;
    size_t BA = (size_t)B * Atot;
    out[(size_t)b * Atot + aidx]      = score;
    out[BA + (size_t)b * Atot + aidx] = (float)bid;
    float* bx = out + 2 * BA + ((size_t)b * Atot + aidx) * 4;
    bx[0] = pcx - 0.5f * pw;
    bx[1] = pcy - 0.5f * ph;
    bx[2] = pcx + 0.5f * pw;
    bx[3] = pcy + 0.5f * ph;
}

// ---------------------------------------------------------------------------
extern "C" void kernel_launch(void* const* d_in, const int* in_sizes, int n_in,
                              void* d_out, int out_size, void* d_ws, size_t ws_size,
                              hipStream_t stream)
{
    const int   HS[5]   = {64, 32, 16, 8, 4};
    const int   LG[5]   = {6, 5, 4, 3, 2};
    const float STRD[5] = {8.f, 16.f, 32.f, 64.f, 128.f};
    const int B = 8;
    const int Atot = (64*64 + 32*32 + 16*16 + 8*8 + 4*4) * 9;   // 49104

    char* ws = (char*)d_ws;
    size_t off = 0;
    auto take = [&](size_t bytes) -> char* {
        char* p = ws + off;
        off = (off + bytes + 255) & ~(size_t)255;
        return p;
    };
    const size_t maxPosPad = (size_t)B * 66 * 66;     // padded level-0 cells
    const size_t maxPos    = (size_t)B * 64 * 64;
    unsigned short* actIn = (unsigned short*)take(maxPosPad * 256 * 2);
    unsigned short* actA  = (unsigned short*)take(maxPosPad * 256 * 2);
    unsigned short* actB  = (unsigned short*)take(maxPosPad * 256 * 2);
    unsigned short* clsO  = (unsigned short*)take(maxPos * 720 * 2);
    unsigned short* regO  = (unsigned short*)take(maxPos * 36 * 2);

    // N-tile-padded channel allocations (multiple of 256 = 16 tiles/block)
    const int clsAllocC[5] = {256, 256, 256, 256, 768};
    const int regAllocC[5] = {256, 256, 256, 256, 256};
    unsigned short* wc[5]; unsigned short* wr[5];
    for (int i = 0; i < 5; ++i) wc[i] = (unsigned short*)take((size_t)clsAllocC[i] * 2304 * 2);
    for (int i = 0; i < 5; ++i) wr[i] = (unsigned short*)take((size_t)regAllocC[i] * 2304 * 2);

    // --- weight repack (10 layers) ---
    for (int i = 0; i < 5; ++i) {
        int co = (i < 4) ? 256 : 720;
        int tc = clsAllocC[i] * 2304;
        wconv_kernel<<<(tc + 255) / 256, 256, 0, stream>>>(
            (const float*)d_in[5 + 2 * i], wc[i], co, clsAllocC[i]);
        int cor = (i < 4) ? 256 : 36;
        int tr = regAllocC[i] * 2304;
        wconv_kernel<<<(tr + 255) / 256, 256, 0, stream>>>(
            (const float*)d_in[15 + 2 * i], wr[i], cor, regAllocC[i]);
    }

    int levelBase = 0;
    for (int l = 0; l < 5; ++l) {
        int H = HS[l], W = HS[l], HW = H * W, logW = LG[l];
        int Hp = H + 2, Wp = W + 2;

        // padded bf16 input with zero halo
        int tc = B * Hp * Wp * 256;
        cvt_kernel<<<(tc + 255) / 256, 256, 0, stream>>>(
            (const float*)d_in[l], actIn, H, W);
        // zero the intermediate buffers' halos (whole-buffer clear, uint4 wide)
        int ndw4 = (B * Hp * Wp * 256) / 8;     // ushorts / 8 per uint4
        zero_kernel<<<(ndw4 + 255) / 256, 256, 0, stream>>>((unsigned int*)actA, ndw4);
        zero_kernel<<<(ndw4 + 255) / 256, 256, 0, stream>>>((unsigned int*)actB, ndw4);

        int numM = (B * HW) / 64;
        auto conv = [&](const unsigned short* src, unsigned short* w2, const float* bias,
                        unsigned short* dst, int Cout, int CoutAlloc, int relu, int padOut) {
            dim3 g(numM, CoutAlloc / 256);
            conv3x3_wmma<<<g, 256, 0, stream>>>(src, w2, bias, dst, H, W, logW,
                                                Cout, relu, padOut);
        };
        // classification tower
        conv(actIn, wc[0], (const float*)d_in[6],  actA, 256, 256, 1, 1);
        conv(actA,  wc[1], (const float*)d_in[8],  actB, 256, 256, 1, 1);
        conv(actB,  wc[2], (const float*)d_in[10], actA, 256, 256, 1, 1);
        conv(actA,  wc[3], (const float*)d_in[12], actB, 256, 256, 1, 1);
        conv(actB,  wc[4], (const float*)d_in[14], clsO, 720, 768, 0, 0);
        // regression tower
        conv(actIn, wr[0], (const float*)d_in[16], actA, 256, 256, 1, 1);
        conv(actA,  wr[1], (const float*)d_in[18], actB, 256, 256, 1, 1);
        conv(actB,  wr[2], (const float*)d_in[20], actA, 256, 256, 1, 1);
        conv(actA,  wr[3], (const float*)d_in[22], actB, 256, 256, 1, 1);
        conv(actB,  wr[4], (const float*)d_in[24], regO, 36, 256, 0, 0);

        int tp = B * HW * 9;
        postproc_kernel<<<(tp + 255) / 256, 256, 0, stream>>>(
            clsO, regO, (float*)d_out, H, W, levelBase, STRD[l], Atot);
        levelBase += HW * 9;
    }
    (void)in_sizes; (void)n_in; (void)out_size; (void)ws_size;
}
